// StreamingTransformer_25434796327767
// MI455X (gfx1250) — matrix operand order repaired and checked by reading, not compiled, not run on hardware
//
#include <hip/hip_runtime.h>
#include <hip/hip_bf16.h>
#include <math.h>

// ---------------------------------------------------------------------------
// StreamingTransformer forward for MI455X (gfx1250, wave32, WMMA).
// GEMMs: v_wmma_f32_16x16x32_f16, 8 WMMAs per wave per k-step (32x64 wave
// tile), all fragments loaded with b128 vector loads (weights pre-transposed).
// ---------------------------------------------------------------------------

#define LL   6
#define DD   512
#define HH   8
#define DFFD 2048
#define WW   256
#define BBv  4
#define TT   1024
#define DKv  64
#define NN   (BBv * TT)          // 4096 token rows
#define PADV (-1.0f)

typedef __attribute__((ext_vector_type(16))) _Float16 v16h;
typedef __attribute__((ext_vector_type(8)))  _Float16 v8h;
typedef __attribute__((ext_vector_type(8)))  float    v8f;

__device__ __forceinline__ float gelu_exact(float x) {
    // torch nn.GELU default (erf variant)
    return 0.5f * x * (1.0f + erff(x * 0.70710678118654752f));
}

__device__ __forceinline__ v16h cat8(v8h lo, v8h hi) {
    return __builtin_shufflevector(lo, hi, 0, 1, 2, 3, 4, 5, 6, 7,
                                           8, 9, 10, 11, 12, 13, 14, 15);
}

// ---------------------------------------------------------------------------
// fp32 -> f16 straight conversion (activations before the heads)
// ---------------------------------------------------------------------------
__global__ void f32_to_f16_kernel(const float* __restrict__ in,
                                  _Float16* __restrict__ out, int n) {
    int i = blockIdx.x * blockDim.x + threadIdx.x;
    if (i < n) out[i] = (_Float16)in[i];
}

// ---------------------------------------------------------------------------
// Weight conversion with transpose: in [K, Kout] f32 -> out [Kout, K] f16.
// Transposed layout makes the WMMA B-fragment (column-per-lane, K along
// halves) a pair of b128 loads from one BT row.
// ---------------------------------------------------------------------------
__global__ void transpose_f16_kernel(const float* __restrict__ in,
                                     _Float16* __restrict__ out,
                                     int K, int Kout) {
    int idx = blockIdx.x * blockDim.x + threadIdx.x;
    if (idx >= K * Kout) return;
    int k = idx / Kout;
    int n = idx - k * Kout;
    out[(size_t)n * K + k] = (_Float16)in[idx];
}

// ---------------------------------------------------------------------------
// h[b,t,:] = x[b,t]*Wp[:] + bp[:] + pos_emb[t,:]
// ---------------------------------------------------------------------------
__global__ void embed_kernel(const float* __restrict__ x,
                             const float* __restrict__ pos,
                             const float* __restrict__ Wp,
                             const float* __restrict__ bp,
                             float* __restrict__ h) {
    int idx = blockIdx.x * blockDim.x + threadIdx.x;   // over NN*DD
    int c   = idx & (DD - 1);
    int row = idx >> 9;                                // /DD
    int t   = row & (TT - 1);
    h[idx] = x[row] * Wp[c] + bp[c] + pos[t * DD + c];
}

// ---------------------------------------------------------------------------
// LayerNorm over D=512, one row per block (128 threads, 4 elems/thread),
// writes f16 normalized activations (GEMM A operand).
// ---------------------------------------------------------------------------
__global__ void ln_kernel(const float* __restrict__ h,
                          const float* __restrict__ g,
                          const float* __restrict__ b,
                          _Float16* __restrict__ out) {
    const int row = blockIdx.x;
    const float* hr = h + (size_t)row * DD;
    __shared__ float red[128];
    float local[4];
    float s = 0.0f;
#pragma unroll
    for (int j = 0; j < 4; ++j) {
        local[j] = hr[threadIdx.x + j * 128];
        s += local[j];
    }
    red[threadIdx.x] = s;
    __syncthreads();
    for (int off = 64; off > 0; off >>= 1) {
        if (threadIdx.x < off) red[threadIdx.x] += red[threadIdx.x + off];
        __syncthreads();
    }
    const float mean = red[0] * (1.0f / DD);
    __syncthreads();
    float v = 0.0f;
#pragma unroll
    for (int j = 0; j < 4; ++j) {
        float d = local[j] - mean;
        v += d * d;
    }
    red[threadIdx.x] = v;
    __syncthreads();
    for (int off = 64; off > 0; off >>= 1) {
        if (threadIdx.x < off) red[threadIdx.x] += red[threadIdx.x + off];
        __syncthreads();
    }
    const float rstd = rsqrtf(red[0] * (1.0f / DD) + 1e-5f);
#pragma unroll
    for (int j = 0; j < 4; ++j) {
        int c = threadIdx.x + j * 128;
        out[(size_t)row * DD + c] =
            (_Float16)((local[j] - mean) * rstd * g[c] + b[c]);
    }
}

// ---------------------------------------------------------------------------
// WMMA GEMM: C[N x Kout] = A[N x K] (f16 row-major) @ W (f16, stored
// TRANSPOSED as BT[Kout x K] row-major) + bias.
//
// Block = 128 threads = 4 waves (2x2) -> 64x128 block tile.
// Each wave owns a 32x64 tile = 2x4 WMMA tiles: per 32-wide k-step it loads
// 2 A-fragments + 4 B-fragments (12 b128 loads) and issues 8 independent
// v_wmma_f32_16x16x32_f16 (1.5 loads per WMMA, deep MFU pipelining).
//
// Fragment layouts per CDNA5 ISA 7.12.2 (wave32):
//   A 16x32: lane(0-15)=row m, halves 0-7 = K k0+g*8..+7, halves 8-15 =
//            K k0+16+g*8..+7  -> two b128 loads per lane.
//   B 32x16: lane = col n, halves 0-7 = K k0+g*8..+7, halves 8-15 =
//            K k0+16+g*8..+7 (BT row-major) -> two b128 loads per lane.
//   C/D v8f: VGPR r -> row r + 8*(lane/16), col = lane%16.
//
// MODE 0: Cf = val            (fp32 store, QKV)
// MODE 1: Ch = gelu(val) f16  (FFN-1 / head hidden)
// MODE 2: Cf += val           (residual accumulate: attn-out proj, FFN-2)
// ---------------------------------------------------------------------------
template <int MODE>
__global__ void gemm_wmma_kernel(const _Float16* __restrict__ A,
                                 const _Float16* __restrict__ BT,
                                 const float* __restrict__ bias,
                                 float* __restrict__ Cf,
                                 _Float16* __restrict__ Ch,
                                 int K, int Kout) {
    const int lane = threadIdx.x & 31;
    const int wave = threadIdx.x >> 5;
    const int row0 = blockIdx.y * 64 + (wave >> 1) * 32;
    const int col0 = blockIdx.x * 128 + (wave & 1) * 64;
    const int g    = lane >> 4;       // lane group (0/1)
    const int mn   = lane & 15;       // m (A rows) / n (B,C cols)

    v8f c[2][4] = {};
    const _Float16* Ar[2];
    Ar[0] = A + (size_t)(row0 + mn) * K;
    Ar[1] = Ar[0] + (size_t)16 * K;
    const _Float16* Br[4];
    Br[0] = BT + (size_t)(col0 + mn) * K;
    Br[1] = Br[0] + (size_t)16 * K;
    Br[2] = Br[0] + (size_t)32 * K;
    Br[3] = Br[0] + (size_t)48 * K;

    for (int k0 = 0; k0 < K; k0 += 32) {
        // A fragments: two b128 loads each (K runs k0+g*8, k0+16+g*8)
        v16h af[2];
#pragma unroll
        for (int mt = 0; mt < 2; ++mt) {
            const v8h* ap = (const v8h*)(Ar[mt] + k0 + g * 8);
            af[mt] = cat8(ap[0], ap[2]);
        }
        // B fragments: two b128 loads each (K runs k0+g*8, k0+16+g*8)
        v16h bf[4];
#pragma unroll
        for (int nt = 0; nt < 4; ++nt) {
            const v8h* bp = (const v8h*)(Br[nt] + k0 + g * 8);
            bf[nt] = cat8(bp[0], bp[2]);
            // Speculative SE-scope prefetch of next k-tile (safe past end).
            __builtin_prefetch((const _Float16*)bp + 32, 0, 3);
        }
#pragma unroll
        for (int mt = 0; mt < 2; ++mt) {
#pragma unroll
            for (int nt = 0; nt < 4; ++nt) {
                c[mt][nt] = __builtin_amdgcn_wmma_f32_16x16x32_f16(
                                false, af[mt], false, bf[nt],
                                (short)0, c[mt][nt], false, false);
            }
        }
    }

    // Epilogue. C/D layout: VGPR r -> row r + 8*laneGroup, col = lane%16.
#pragma unroll
    for (int nt = 0; nt < 4; ++nt) {
        const int col = col0 + nt * 16 + mn;
        const float bb = bias[col];
#pragma unroll
        for (int mt = 0; mt < 2; ++mt) {
#pragma unroll
            for (int r = 0; r < 8; ++r) {
                int row = row0 + mt * 16 + g * 8 + r;
                float val = c[mt][nt][r] + bb;
                size_t idx = (size_t)row * Kout + col;
                if (MODE == 0)      Cf[idx] = val;
                else if (MODE == 1) Ch[idx] = (_Float16)gelu_exact(val);
                else                Cf[idx] += val;
            }
        }
    }
}

// ---------------------------------------------------------------------------
// Sliding-window causal attention with key-padding mask, online softmax.
// One thread per (b, head, query). DK=64 q and accumulator in VGPRs as
// float4; k/v rows streamed with b128 loads.
// ---------------------------------------------------------------------------
__global__ void attn_kernel(const float* __restrict__ q,
                            const float* __restrict__ k,
                            const float* __restrict__ v,
                            const float* __restrict__ x,
                            _Float16* __restrict__ o) {
    int idx = blockIdx.x * blockDim.x + threadIdx.x;   // [0, B*H*T)
    int i   = idx & (TT - 1);
    int bh  = idx >> 10;
    int hh  = bh & (HH - 1);
    int b   = bh >> 3;

    const size_t qoff = ((size_t)(b * TT + i) * DD) + hh * DKv;
    const float4* q4 = (const float4*)(q + qoff);
    float4 qr[DKv / 4], acc[DKv / 4];
#pragma unroll
    for (int d = 0; d < DKv / 4; ++d) {
        qr[d]  = q4[d];
        acc[d] = make_float4(0.f, 0.f, 0.f, 0.f);
    }

    float m = -INFINITY, l = 0.0f;
    int jlo = i - (WW - 1);
    if (jlo < 0) jlo = 0;
    for (int j = jlo; j <= i; ++j) {
        if (x[b * TT + j] == PADV) continue;           // key padding -> -inf
        const size_t koff = ((size_t)(b * TT + j) * DD) + hh * DKv;
        const float4* kr = (const float4*)(k + koff);
        float s = 0.0f;
#pragma unroll
        for (int d = 0; d < DKv / 4; ++d) {
            float4 kv = kr[d];
            s += qr[d].x * kv.x + qr[d].y * kv.y + qr[d].z * kv.z + qr[d].w * kv.w;
        }
        s *= 0.125f;                                   // 1/sqrt(64)
        float mnew = fmaxf(m, s);
        float corr = __expf(m - mnew);
        float p    = __expf(s - mnew);
        const float4* vr = (const float4*)(v + koff);
        l = l * corr + p;
#pragma unroll
        for (int d = 0; d < DKv / 4; ++d) {
            float4 vv = vr[d];
            acc[d].x = acc[d].x * corr + p * vv.x;
            acc[d].y = acc[d].y * corr + p * vv.y;
            acc[d].z = acc[d].z * corr + p * vv.z;
            acc[d].w = acc[d].w * corr + p * vv.w;
        }
        m = mnew;
    }
    float inv = 1.0f / l;
#pragma unroll
    for (int d = 0; d < DKv / 4; ++d) {
        o[qoff + 4 * d + 0] = (_Float16)(acc[d].x * inv);
        o[qoff + 4 * d + 1] = (_Float16)(acc[d].y * inv);
        o[qoff + 4 * d + 2] = (_Float16)(acc[d].z * inv);
        o[qoff + 4 * d + 3] = (_Float16)(acc[d].w * inv);
    }
}

// ---------------------------------------------------------------------------
// Final tiny head projections (Kout = 2/1/1): plain dot products.
// out layout: binary [0,8192), anomaly [8192,12288), caustic [12288,16384).
// ---------------------------------------------------------------------------
__global__ void heads_kernel(const _Float16* __restrict__ t1,   // [N,256]
                             const _Float16* __restrict__ t2,   // [N,128]
                             const _Float16* __restrict__ t3,   // [N,128]
                             const float* __restrict__ bhW2,    // [256,2]
                             const float* __restrict__ bhb2,
                             const float* __restrict__ ahW2,    // [128,1]
                             const float* __restrict__ ahb2,
                             const float* __restrict__ chW2,    // [128,1]
                             const float* __restrict__ chb2,
                             float* __restrict__ out) {
    int n = blockIdx.x * blockDim.x + threadIdx.x;     // [0, NN)
    const _Float16* r1 = t1 + (size_t)n * 256;
    float s0 = bhb2[0], s1 = bhb2[1];
#pragma unroll 4
    for (int j = 0; j < 256; ++j) {
        float tv = (float)r1[j];
        s0 += tv * bhW2[j * 2 + 0];
        s1 += tv * bhW2[j * 2 + 1];
    }
    out[n * 2 + 0] = s0;
    out[n * 2 + 1] = s1;

    const _Float16* r2 = t2 + (size_t)n * 128;
    float sa = ahb2[0];
#pragma unroll 4
    for (int j = 0; j < 128; ++j) sa += (float)r2[j] * ahW2[j];
    out[2 * NN + n] = sa;

    const _Float16* r3 = t3 + (size_t)n * 128;
    float sc = chb2[0];
#pragma unroll 4
    for (int j = 0; j < 128; ++j) sc += (float)r3[j] * chW2[j];
    out[3 * NN + n] = 1.0f / (1.0f + __expf(-sc));
}

// ---------------------------------------------------------------------------
// Host-side orchestration
// ---------------------------------------------------------------------------
extern "C" void kernel_launch(void* const* d_in, const int* in_sizes, int n_in,
                              void* d_out, int out_size, void* d_ws, size_t ws_size,
                              hipStream_t stream) {
    (void)in_sizes; (void)n_in; (void)out_size; (void)ws_size;

    const float* x      = (const float*)d_in[0];
    const float* pos    = (const float*)d_in[1];
    const float* Wp     = (const float*)d_in[2];
    const float* bp     = (const float*)d_in[3];
    const float* Wq     = (const float*)d_in[4];
    const float* bq     = (const float*)d_in[5];
    const float* Wk     = (const float*)d_in[6];
    const float* bk     = (const float*)d_in[7];
    const float* Wv     = (const float*)d_in[8];
    const float* bv     = (const float*)d_in[9];
    const float* Wo     = (const float*)d_in[10];
    const float* bo     = (const float*)d_in[11];
    const float* ln1_g  = (const float*)d_in[12];
    const float* ln1_b  = (const float*)d_in[13];
    const float* W1     = (const float*)d_in[14];
    const float* b1     = (const float*)d_in[15];
    const float* W2     = (const float*)d_in[16];
    const float* b2     = (const float*)d_in[17];
    const float* ln2_g  = (const float*)d_in[18];
    const float* ln2_b  = (const float*)d_in[19];
    const float* bhW1   = (const float*)d_in[20];
    const float* bhb1   = (const float*)d_in[21];
    const float* bhW2   = (const float*)d_in[22];
    const float* bhb2   = (const float*)d_in[23];
    const float* ahW1   = (const float*)d_in[24];
    const float* ahb1   = (const float*)d_in[25];
    const float* ahW2   = (const float*)d_in[26];
    const float* ahb2   = (const float*)d_in[27];
    const float* chW1   = (const float*)d_in[28];
    const float* chb1   = (const float*)d_in[29];
    const float* chW2   = (const float*)d_in[30];
    const float* chb2   = (const float*)d_in[31];
    float* out = (float*)d_out;

    // Workspace carving (256B aligned)
    char* ws = (char*)d_ws;
    size_t off = 0;
    auto carve = [&](size_t bytes) -> char* {
        char* p = ws + off;
        off += (bytes + 255) & ~(size_t)255;
        return p;
    };
    // Transposed f16 weights: [Kout, K] row-major per layer
    _Float16* wq16   = (_Float16*)carve((size_t)LL * DD * DD * 2);
    _Float16* wk16   = (_Float16*)carve((size_t)LL * DD * DD * 2);
    _Float16* wv16   = (_Float16*)carve((size_t)LL * DD * DD * 2);
    _Float16* wo16   = (_Float16*)carve((size_t)LL * DD * DD * 2);
    _Float16* w1_16  = (_Float16*)carve((size_t)LL * DD * DFFD * 2);
    _Float16* w2_16  = (_Float16*)carve((size_t)LL * DFFD * DD * 2);
    _Float16* bhW116 = (_Float16*)carve((size_t)DD * 256 * 2);
    _Float16* ahW116 = (_Float16*)carve((size_t)DD * 128 * 2);
    _Float16* chW116 = (_Float16*)carve((size_t)DD * 128 * 2);
    float*    h      = (float*)   carve((size_t)NN * DD * 4);
    _Float16* hn16   = (_Float16*)carve((size_t)NN * DD * 2);
    float*    q32    = (float*)   carve((size_t)NN * DD * 4);
    float*    k32    = (float*)   carve((size_t)NN * DD * 4);
    float*    v32    = (float*)   carve((size_t)NN * DD * 4);
    _Float16* o16    = (_Float16*)carve((size_t)NN * DD * 2);
    _Float16* ff16   = (_Float16*)carve((size_t)NN * DFFD * 2);
    _Float16* h16    = (_Float16*)carve((size_t)NN * DD * 2);
    _Float16* t1_16  = (_Float16*)carve((size_t)NN * 256 * 2);
    _Float16* t2_16  = (_Float16*)carve((size_t)NN * 128 * 2);
    _Float16* t3_16  = (_Float16*)carve((size_t)NN * 128 * 2);

    auto cvtT = [&](const float* src, _Float16* dst, int K, int Kout) {
        int n = K * Kout;
        transpose_f16_kernel<<<(n + 255) / 256, 256, 0, stream>>>(src, dst, K, Kout);
    };

    // Convert+transpose all GEMM weights to f16 once per call (deterministic).
    for (int l = 0; l < LL; ++l) {
        cvtT(Wq + (size_t)l * DD * DD,   wq16  + (size_t)l * DD * DD,   DD,   DD);
        cvtT(Wk + (size_t)l * DD * DD,   wk16  + (size_t)l * DD * DD,   DD,   DD);
        cvtT(Wv + (size_t)l * DD * DD,   wv16  + (size_t)l * DD * DD,   DD,   DD);
        cvtT(Wo + (size_t)l * DD * DD,   wo16  + (size_t)l * DD * DD,   DD,   DD);
        cvtT(W1 + (size_t)l * DD * DFFD, w1_16 + (size_t)l * DD * DFFD, DD,   DFFD);
        cvtT(W2 + (size_t)l * DFFD * DD, w2_16 + (size_t)l * DFFD * DD, DFFD, DD);
    }
    cvtT(bhW1, bhW116, DD, 256);
    cvtT(ahW1, ahW116, DD, 128);
    cvtT(chW1, chW116, DD, 128);

    // Input projection + positional embedding
    embed_kernel<<<(NN * DD) / 256, 256, 0, stream>>>(x, pos, Wp, bp, h);

    const dim3 gemmBlk(128);
    for (int l = 0; l < LL; ++l) {
        const size_t wofD = (size_t)l * DD * DD;
        const size_t wofF = (size_t)l * DD * DFFD;

        // ---- attention sub-block ----
        ln_kernel<<<NN, 128, 0, stream>>>(h, ln1_g + l * DD, ln1_b + l * DD, hn16);
        gemm_wmma_kernel<0><<<dim3(DD / 128, NN / 64), gemmBlk, 0, stream>>>(
            hn16, wq16 + wofD, bq + l * DD, q32, nullptr, DD, DD);
        gemm_wmma_kernel<0><<<dim3(DD / 128, NN / 64), gemmBlk, 0, stream>>>(
            hn16, wk16 + wofD, bk + l * DD, k32, nullptr, DD, DD);
        gemm_wmma_kernel<0><<<dim3(DD / 128, NN / 64), gemmBlk, 0, stream>>>(
            hn16, wv16 + wofD, bv + l * DD, v32, nullptr, DD, DD);
        attn_kernel<<<(BBv * HH * TT) / 128, 128, 0, stream>>>(q32, k32, v32, x, o16);
        gemm_wmma_kernel<2><<<dim3(DD / 128, NN / 64), gemmBlk, 0, stream>>>(
            o16, wo16 + wofD, bo + l * DD, h, nullptr, DD, DD);   // h += o@Wo+bo

        // ---- FFN sub-block ----
        ln_kernel<<<NN, 128, 0, stream>>>(h, ln2_g + l * DD, ln2_b + l * DD, hn16);
        gemm_wmma_kernel<1><<<dim3(DFFD / 128, NN / 64), gemmBlk, 0, stream>>>(
            hn16, w1_16 + wofF, b1 + l * DFFD, nullptr, ff16, DD, DFFD); // gelu,f16
        gemm_wmma_kernel<2><<<dim3(DD / 128, NN / 64), gemmBlk, 0, stream>>>(
            ff16, w2_16 + wofF, b2 + l * DD, h, nullptr, DFFD, DD);      // h += ...
    }

    // ---- output heads ----
    f32_to_f16_kernel<<<(NN * DD) / 256, 256, 0, stream>>>(h, h16, NN * DD);
    gemm_wmma_kernel<1><<<dim3(256 / 128, NN / 64), gemmBlk, 0, stream>>>(
        h16, bhW116, bhb1, nullptr, t1_16, DD, 256);
    gemm_wmma_kernel<1><<<dim3(128 / 128, NN / 64), gemmBlk, 0, stream>>>(
        h16, ahW116, ahb1, nullptr, t2_16, DD, 128);
    gemm_wmma_kernel<1><<<dim3(128 / 128, NN / 64), gemmBlk, 0, stream>>>(
        h16, chW116, chb1, nullptr, t3_16, DD, 128);
    heads_kernel<<<NN / 128, 128, 0, stream>>>(
        t1_16, t2_16, t3_16, bhW2, bhb2, ahW2, ahb2, chW2, chb2, out);
}
